// Transformer_64785286693620
// MI455X (gfx1250) — compile-verified, matrix operand order checked by reading, not run on hardware
//
#include <hip/hip_runtime.h>
#include <hip/hip_bf16.h>
#include <cstdint>

// ---------------- types for CDNA5 WMMA / TDM ----------------
typedef __bf16 v16bf __attribute__((ext_vector_type(16)));
typedef float  v8f   __attribute__((ext_vector_type(8)));
typedef unsigned int u32x4 __attribute__((ext_vector_type(4)));
typedef int          i32x8 __attribute__((ext_vector_type(8)));
typedef int          i32x4 __attribute__((ext_vector_type(4)));

#define N_NODES 20000
#define E_EDGES 320000
#define IN_DIMC 512
#define HIDC    256
#define NHEAD   8
#define DHC     32

// ---------------- small elementwise kernels ----------------
__global__ void f2bf_kernel(const float* __restrict__ in, __bf16* __restrict__ out, int n) {
    int i = blockIdx.x * 256 + threadIdx.x;
    if (i < n) out[i] = (__bf16)in[i];
}

__global__ void zero_kernel(float* __restrict__ p, int n) {
    int i = blockIdx.x * 256 + threadIdx.x;
    if (i < n) p[i] = 0.0f;
}

__global__ void copy_kernel(const float* __restrict__ in, float* __restrict__ out, int n) {
    int i = blockIdx.x * 256 + threadIdx.x;
    if (i < n) out[i] = in[i];
}

// Pre-swizzle weight W[K,N] (f32) into WMMA-B-fragment-major bf16 layout:
// tile t = (kk/32)*(N/16) + n0/16 owns 512 contiguous bf16; within a tile,
// lane L's 16 fragment elements are contiguous at [t*512 + L*16, ...+16).
// Element e=2i+j of lane L corresponds to B row k = pat(i) + 8*(L>=16) + j,
// col = n0 + (L&15), matching the 16-bit WMMA B VGPR layout.
__global__ void wswizzle_kernel(const float* __restrict__ Wf, __bf16* __restrict__ out,
                                int K, int N) {
    int idx = blockIdx.x * 256 + threadIdx.x;
    if (idx >= K * N) return;
    int tile = idx >> 9, within = idx & 511;
    int lane = within >> 4, e = within & 15;
    int i = e >> 1, j = e & 1;
    int kb = (lane >= 16) ? 8 : 0;
    int krow = ((i < 4) ? 2 * i : 16 + 2 * (i - 4)) + kb + j;
    int ntiles = N >> 4;
    int tileK = tile / ntiles, tileN = tile - tileK * ntiles;
    out[idx] = (__bf16)Wf[(size_t)(tileK * 32 + krow) * N + tileN * 16 + (lane & 15)];
}

// attn = wV / (z + 1e-6), converted to bf16 for the Wo GEMM
__global__ void norm_convert_kernel(const float* __restrict__ wV, const float* __restrict__ z,
                                    __bf16* __restrict__ out, int n) {
    int i = blockIdx.x * 256 + threadIdx.x;
    if (i >= n) return;
    int row = i >> 8, c = i & 255;
    out[i] = (__bf16)(wV[i] / (z[row * NHEAD + (c >> 5)] + 1e-6f));
}

// residual + layernorm, in place on h. One 256-thread block per row.
__global__ __launch_bounds__(256) void add_ln_kernel(float* __restrict__ h,
                                                     const float* __restrict__ delta,
                                                     const float* __restrict__ g,
                                                     const float* __restrict__ b) {
    __shared__ float red[256];
    int row = blockIdx.x, c = threadIdx.x;
    float v = h[row * HIDC + c] + delta[row * HIDC + c];
    red[c] = v; __syncthreads();
    for (int s = 128; s > 0; s >>= 1) { if (c < s) red[c] += red[c + s]; __syncthreads(); }
    float mean = red[0] * (1.0f / HIDC);
    __syncthreads();
    float dv = v - mean;
    red[c] = dv * dv; __syncthreads();
    for (int s = 128; s > 0; s >>= 1) { if (c < s) red[c] += red[c + s]; __syncthreads(); }
    float var = red[0] * (1.0f / HIDC);
    h[row * HIDC + c] = dv * rsqrtf(var + 1e-5f) * g[c] + b[c];
}

// ---------------- per-layer low-rank precompute ----------------
__global__ __launch_bounds__(256) void layer_pre_kernel(
    const float* __restrict__ esw, const float* __restrict__ esb,
    const float* __restrict__ Wsim, const float* __restrict__ bsim,
    const float* __restrict__ epw, const float* __restrict__ epb,
    const float* __restrict__ Wphi, const float* __restrict__ bphi,
    float* __restrict__ simM, float* __restrict__ simdv,
    float* __restrict__ headA, float* __restrict__ headC) {
    int j = threadIdx.x;
    for (int t = 0; t < 8; t++) {
        float acc = 0.f;
        for (int c = 0; c < HIDC; c++) acc += esw[t * HIDC + c] * Wsim[c * HIDC + j];
        simM[t * HIDC + j] = acc;
    }
    float acc = 0.f;
    for (int c = 0; c < HIDC; c++) acc += esb[c] * Wsim[c * HIDC + j];
    simdv[j] = acc + bsim[j];
    float a = 0.f, cc = 0.f;
    for (int c = 0; c < HIDC; c++) {
        float w = Wphi[c * HIDC + j];
        a  += epw[c] * w;
        cc += epb[c] * w;
    }
    cc += bphi[j];
    __shared__ float sa[256], sc[256];
    sa[j] = a; sc[j] = cc; __syncthreads();
    if (j < NHEAD) {
        float A = 0.f, C = 0.f;
        for (int t = 0; t < DHC; t++) { A += sa[j * DHC + t]; C += sc[j * DHC + t]; }
        headA[j] = A; headC[j] = C;
    }
}

// ---------------- edge attention (gather / score / scatter) ----------------
__global__ __launch_bounds__(256) void edge_attn_kernel(
    const float* __restrict__ Q, const float* __restrict__ Km, const float* __restrict__ V,
    const float* __restrict__ esim, const float* __restrict__ ephi,
    const int* __restrict__ src, const int* __restrict__ dst,
    const float* __restrict__ simM, const float* __restrict__ simdv,
    const float* __restrict__ headA, const float* __restrict__ headC,
    float* __restrict__ wV, float* __restrict__ z, int E) {
    int wid  = (blockIdx.x * 256 + threadIdx.x) >> 5;
    int lane = threadIdx.x & 31;
    if (wid >= E) return;
    int s = src[wid], d = dst[wid];
    float es[8];
#pragma unroll
    for (int t = 0; t < 8; t++) es[t] = esim[wid * 8 + t];
    float part = 0.f;
#pragma unroll
    for (int i = 0; i < 8; i++) {
        int c = lane * 8 + i;
        float acc = simdv[c];
#pragma unroll
        for (int t = 0; t < 8; t++) acc += es[t] * simM[t * HIDC + c];
        part += (Km[s * HIDC + c] + acc) * (Q[d * HIDC + c] + acc);
    }
    part += __shfl_xor(part, 1, 32);
    part += __shfl_xor(part, 2, 32);   // dot over the 32-dim head (4 lanes)
    int hd = lane >> 2;
    float sc = part * 5.656854249492380f + ephi[wid] * headA[hd] + headC[hd];
    sc = fminf(fmaxf(sc, -8.f), 8.f);
    float w = __expf(sc);
#pragma unroll
    for (int i = 0; i < 8; i++) {
        int c = lane * 8 + i;
        atomicAdd(&wV[d * HIDC + c], w * V[s * HIDC + c]);
    }
    if ((lane & 3) == 0) atomicAdd(&z[d * NHEAD + hd], w);
}

// ---------------- TDM tile stage: 16 rows x 32 cols of bf16 into LDS ----------------
__device__ __forceinline__ void tdm_issue_tile(const __bf16* gsrc, __bf16* ldst,
                                               int K, int rowsLeft) {
#if __has_builtin(__builtin_amdgcn_tensor_load_to_lds)
    unsigned long long ga = (unsigned long long)(size_t)gsrc;
    unsigned ldsOff = (unsigned)(size_t)(void*)ldst;             // LDS offset = addr[31:0]
    u32x4 g0;
    g0.x = 1u;                                                   // count = 1 valid descriptor
    g0.y = ldsOff;                                               // lds_addr
    g0.z = (unsigned)ga;                                         // global_addr[31:0]
    g0.w = (unsigned)((ga >> 32) & 0x1ffffffull) | (2u << 30);   // addr[56:32] | type=2
    unsigned long long tdim0 = (unsigned)K;                      // tensor dim0 (elements)
    unsigned long long tdim1 = (unsigned)rowsLeft;               // rows remaining
    unsigned long long q0 = (1ull << 16) | (tdim0 << 48);                  // data_size=2B
    unsigned long long q1 = (tdim0 >> 16) | (tdim1 << 16) | (32ull << 48); // tile_dim0=32
    unsigned long long q2 = 16ull | ((tdim0 & 0xffffffffull) << 32);       // tile_dim1=16, stride0=K
    unsigned long long q3 = (tdim0 >> 32);
    i32x8 g1;
    g1[0] = (int)q0; g1[1] = (int)(q0 >> 32);
    g1[2] = (int)q1; g1[3] = (int)(q1 >> 32);
    g1[4] = (int)q2; g1[5] = (int)(q2 >> 32);
    g1[6] = (int)q3; g1[7] = (int)(q3 >> 32);
    i32x4 gz4 = {0, 0, 0, 0};
    i32x8 gz8 = {0, 0, 0, 0, 0, 0, 0, 0};
    __builtin_amdgcn_tensor_load_to_lds(g0, g1, gz4, gz4, gz8, 0);
#else
    int lane = threadIdx.x & 31;
    for (int e = lane; e < 16 * 32; e += 32) ldst[e] = gsrc[(size_t)(e >> 5) * K + (e & 31)];
#endif
}

__device__ __forceinline__ void tdm_wait0() {
#if __has_builtin(__builtin_amdgcn_tensor_load_to_lds)
    __builtin_amdgcn_s_wait_tensorcnt((short)0);
#endif
}

// ---------------- bf16 WMMA GEMM ----------------
// C[M,N] = A[M,K] @ W_swizzled[K,N] (+bias) (+act). Block = 8 waves, tile 16 x 128.
// A tile double-buffered in LDS via the Tensor Data Mover (DMA overlapped with WMMA);
// B fragments are single 32B vector loads from the pre-swizzled weight.
// act: 0 = none, 1 = relu, 2 = selu
__global__ __launch_bounds__(256) void gemm_bf16_kernel(
    const __bf16* __restrict__ A, const __bf16* __restrict__ W,
    const float* __restrict__ bias, float* __restrict__ C,
    int M, int K, int N, int act) {
    __shared__ __attribute__((aligned(64))) __bf16 lA[2][16 * 32];
    int tid  = threadIdx.x;
    int wave = tid >> 5, lane = tid & 31;
    int m0 = blockIdx.x * 16;
    int n0 = blockIdx.y * 128 + wave * 16;
    int row = lane & 15;
    int hiK = (lane >> 4) * 8;
    int ntilesN = N >> 4;
    v8f acc = {};

    if (wave == 0) {
        tdm_issue_tile(A + (size_t)m0 * K, &lA[0][0], K, M - m0);
        tdm_wait0();
    }
    __syncthreads();

    int cur = 0;
    for (int kk = 0; kk < K; kk += 32) {
        // kick off DMA for the next K-step while this one computes
        if (kk + 32 < K && wave == 0)
            tdm_issue_tile(A + (size_t)m0 * K + (kk + 32), &lA[cur ^ 1][0], K, M - m0);

        v16bf a;
#pragma unroll
        for (int i = 0; i < 8; i++) {
            int k = ((i < 4) ? (2 * i) : (16 + 2 * (i - 4))) + hiK;
            a[2 * i]     = lA[cur][row * 32 + k];
            a[2 * i + 1] = lA[cur][row * 32 + k + 1];
        }
        v16bf b = *(const v16bf*)(W + ((size_t)(kk >> 5) * ntilesN + (n0 >> 4)) * 512
                                    + lane * 16);
        acc = __builtin_amdgcn_wmma_f32_16x16x32_bf16(false, a, false, b,
                                                      (short)0, acc, false, false);
        if (kk + 32 < K && wave == 0) tdm_wait0();
        __syncthreads();
        cur ^= 1;
    }

    int col   = n0 + row;
    int rbase = m0 + ((lane >> 4) << 3);
#pragma unroll
    for (int r = 0; r < 8; r++) {
        float v = acc[r] + (bias ? bias[col] : 0.f);
        if (act == 1)      v = fmaxf(v, 0.f);
        else if (act == 2) v = v > 0.f ? 1.0507009873554805f * v
                                       : 1.0507009873554805f * 1.6732632423543772f * (__expf(v) - 1.f);
        C[(size_t)(rbase + r) * N + col] = v;
    }
}

// ---------------- host orchestration ----------------
extern "C" void kernel_launch(void* const* d_in, const int* in_sizes, int n_in,
                              void* d_out, int out_size, void* d_ws, size_t ws_size,
                              hipStream_t stream) {
    (void)in_sizes; (void)n_in; (void)out_size; (void)ws_size;
    // input order: x, edge_phi, edge_sim, params{...}, edge_src, edge_dst
    const float* x         = (const float*)d_in[0];
    const float* edge_phi  = (const float*)d_in[1];
    const float* edge_sim  = (const float*)d_in[2];
    const float* emb_h_w   = (const float*)d_in[3];
    const float* emb_phi_w = (const float*)d_in[4];
    const float* emb_phi_b = (const float*)d_in[5];
    const float* emb_sim_w = (const float*)d_in[6];
    const float* emb_sim_b = (const float*)d_in[7];
    const float* mlp_w1    = (const float*)d_in[8];
    const float* mlp_b1    = (const float*)d_in[9];
    const float* mlp_w2    = (const float*)d_in[10];
    const float* mlp_b2    = (const float*)d_in[11];
    const float* L[2][20];
    for (int l = 0; l < 2; l++)
        for (int t = 0; t < 20; t++) L[l][t] = (const float*)d_in[12 + 20 * l + t];
    // layer params: 0 Wq 1 bq 2 Wk 3 bk 4 Wv 5 bv 6 Wsim 7 bsim 8 Wphi 9 bphi
    //               10 Wo 11 bo 12 Wf1 13 bf1 14 Wf2 15 bf2 16 ln1_g 17 ln1_b 18 ln2_g 19 ln2_b
    const int* src = (const int*)d_in[52];
    const int* dst = (const int*)d_in[53];
    float* out = (float*)d_out;

    // workspace layout (f32 region, then 64B-aligned bf16 region)
    float* h    = (float*)d_ws;                     // N*256
    float* Q    = h    + (size_t)N_NODES * HIDC;
    float* Km   = Q    + (size_t)N_NODES * HIDC;
    float* V    = Km   + (size_t)N_NODES * HIDC;
    float* h2   = V    + (size_t)N_NODES * HIDC;    // gemm outputs / residual delta
    float* wV   = h2   + (size_t)N_NODES * HIDC;    // attention numerator
    float* z    = wV   + (size_t)N_NODES * HIDC;    // N*8 denominators
    float* ffh  = z    + (size_t)N_NODES * NHEAD;   // N*512 ffn hidden
    float* simM = ffh  + (size_t)N_NODES * IN_DIMC; // 8*256
    float* simd = simM + 8 * HIDC;                  // 256
    float* hA   = simd + HIDC;                      // 8
    float* hC   = hA + NHEAD;                       // 8
    size_t fbytes = (size_t)((char*)(hC + NHEAD) - (char*)d_ws);
    fbytes = (fbytes + 63) & ~(size_t)63;
    __bf16* Abf = (__bf16*)((char*)d_ws + fbytes);  // N*512 bf16 (64B aligned)
    __bf16* Wbf = Abf + (size_t)N_NODES * IN_DIMC;  // 512*512 bf16 (64B aligned)

    auto cvt = [&](const float* in, __bf16* o, int n) {
        f2bf_kernel<<<(n + 255) / 256, 256, 0, stream>>>(in, o, n);
    };
    auto gemm = [&](const __bf16* A, const float* Wf, const float* bias, float* C,
                    int M, int K, int N, int act) {
        wswizzle_kernel<<<((K * N) + 255) / 256, 256, 0, stream>>>(Wf, Wbf, K, N);
        dim3 g(M / 16, N / 128);
        gemm_bf16_kernel<<<g, 256, 0, stream>>>(A, Wbf, bias, C, M, K, N, act);
    };

    // ---- embedding: h = x @ emb_h_w ----
    cvt(x, Abf, N_NODES * IN_DIMC);
    gemm(Abf, emb_h_w, nullptr, h, N_NODES, IN_DIMC, HIDC, 0);

    // ---- transformer layers ----
    for (int l = 0; l < 2; l++) {
        layer_pre_kernel<<<1, 256, 0, stream>>>(emb_sim_w, emb_sim_b, L[l][6], L[l][7],
                                                emb_phi_w, emb_phi_b, L[l][8], L[l][9],
                                                simM, simd, hA, hC);
        cvt(h, Abf, N_NODES * HIDC);
        gemm(Abf, L[l][0], L[l][1], Q,  N_NODES, HIDC, HIDC, 0);
        gemm(Abf, L[l][2], L[l][3], Km, N_NODES, HIDC, HIDC, 0);
        gemm(Abf, L[l][4], L[l][5], V,  N_NODES, HIDC, HIDC, 0);

        int nz = N_NODES * (HIDC + NHEAD);          // wV and z are contiguous
        zero_kernel<<<(nz + 255) / 256, 256, 0, stream>>>(wV, nz);
        edge_attn_kernel<<<E_EDGES / 8, 256, 0, stream>>>(Q, Km, V, edge_sim, edge_phi,
                                                          src, dst, simM, simd, hA, hC,
                                                          wV, z, E_EDGES);

        norm_convert_kernel<<<(N_NODES * HIDC + 255) / 256, 256, 0, stream>>>(
            wV, z, Abf, N_NODES * HIDC);
        gemm(Abf, L[l][10], L[l][11], h2, N_NODES, HIDC, HIDC, 0);
        add_ln_kernel<<<N_NODES, 256, 0, stream>>>(h, h2, L[l][16], L[l][17]);

        cvt(h, Abf, N_NODES * HIDC);
        gemm(Abf, L[l][12], L[l][13], ffh, N_NODES, HIDC, 2 * HIDC, 1);   // relu
        cvt(ffh, Abf, N_NODES * 2 * HIDC);
        gemm(Abf, L[l][14], L[l][15], h2, N_NODES, 2 * HIDC, HIDC, 0);
        add_ln_kernel<<<N_NODES, 256, 0, stream>>>(h, h2, L[l][18], L[l][19]);
    }

    // ---- reconstruction head: x_hat = selu(h@w1+b1)@w2+b2 ----
    cvt(h, Abf, N_NODES * HIDC);
    gemm(Abf, mlp_w1, mlp_b1, h2, N_NODES, HIDC, HIDC, 2);                // selu
    cvt(h2, Abf, N_NODES * HIDC);
    gemm(Abf, mlp_w2, mlp_b2, out + (size_t)N_NODES * HIDC, N_NODES, HIDC, IN_DIMC, 0);

    // ---- outputs: (h, x_hat) concatenated ----
    copy_kernel<<<(N_NODES * HIDC + 255) / 256, 256, 0, stream>>>(h, out, N_NODES * HIDC);
}
